// Block_41188736368834
// MI455X (gfx1250) — matrix operand order repaired and checked by reading, not compiled, not run on hardware
//
#include <hip/hip_runtime.h>
#include <stdint.h>

// ---------------- problem constants (match reference) ----------------
#define B_DIM   2
#define T_SEQ   2048
#define C_DIM   1024
#define H_NUM   16
#define D_DIM   64
#define HID_DIM 4096
#define M_ROWS  (B_DIM * T_SEQ)   // 4096 rows of activations
#define LN_EPS  1e-5f

// ---------------- CDNA5 WMMA types ----------------
typedef __attribute__((ext_vector_type(4)))  unsigned int v4u;
typedef __attribute__((ext_vector_type(8)))  unsigned int v8u;
typedef __attribute__((ext_vector_type(16))) __bf16       v16bf;
typedef __attribute__((ext_vector_type(8)))  float        v8f;

union FragBF { v4u q[2]; v16bf v; };

__device__ __forceinline__ unsigned short f2bf(float f) {
  unsigned int u = __float_as_uint(f);
  u += 0x7FFFu + ((u >> 16) & 1u);          // round-to-nearest-even
  return (unsigned short)(u >> 16);
}

// ---------------- fp32 -> bf16 conversion ----------------
__global__ void cvt_kernel(const float* __restrict__ in,
                           unsigned short* __restrict__ out, int n) {
  int i = blockIdx.x * blockDim.x + threadIdx.x;
  int stride = gridDim.x * blockDim.x;
  for (; i < n; i += stride) out[i] = f2bf(in[i]);
}

// ---------------- bf16 WMMA GEMM with fused epilogue ----------------
// C[M,N] = act( A[M,K](bf16) @ B[K,N](bf16) + bias ) + residual
// A tile (128x32) is staged by the Tensor Data Mover (tensor_load_to_lds)
// with hardware LDS padding producing the 40-element row pitch.
#define GBM 128
#define GBN 64
#define GBK 32
#define APITCH 40   // bf16 elements per LDS row (32 + 8 pad, from TDM pad)

template <int RESID, int HASF, int HASB, int GELU>
__global__ __launch_bounds__(256) void gemm_bf16_kernel(
    const unsigned short* __restrict__ A,
    const unsigned short* __restrict__ Bw,
    int M, int N, int K,
    const float* __restrict__ bias,
    const float* __restrict__ residual,
    float* __restrict__ outF,
    unsigned short* __restrict__ outB) {
  __shared__ unsigned short As[GBM * APITCH];  // [m][k], pitch 40
  __shared__ unsigned short Bs[GBN * APITCH];  // [n][k] (transposed)

  const int tid  = threadIdx.x;
  const int lane = tid & 31;
  const int wave = tid >> 5;     // 8 waves
  const int wm   = wave >> 1;    // 0..3  (32-row slab)
  const int wn   = wave & 1;     // 0..1  (32-col slab)
  const int bm   = blockIdx.y * GBM;
  const int bn   = blockIdx.x * GBN;
  const int l15  = lane & 15;
  const int lh   = lane >> 4;    // lane half

  // LDS byte offset of As (flat LDS addresses truncate to the LDS offset)
  const unsigned lds_base = (unsigned)(uintptr_t)(void*)As;

  v8f acc[2][2];
#pragma unroll
  for (int i = 0; i < 2; ++i)
#pragma unroll
    for (int j = 0; j < 2; ++j)
#pragma unroll
      for (int e = 0; e < 8; ++e) acc[i][j][e] = 0.0f;

  for (int k0 = 0; k0 < K; k0 += GBK) {
    // ---- stage A tile 128x32 via TDM (wave 0 issues, EXEC-independent) ----
    if (wave == 0) {
      unsigned long long ga =
          (unsigned long long)(uintptr_t)(A + (size_t)bm * K + k0);
      v4u g0;
      g0[0] = 1u;                                   // count=1, user mode
      g0[1] = lds_base;                             // lds_addr
      g0[2] = (unsigned)ga;                         // global_addr[31:0]
      g0[3] = (unsigned)(ga >> 32) | (2u << 30);    // addr[56:32] | type=2
      v8u g1;
      g1[0] = (1u << 16)      // data_size = 2 bytes
            | (1u << 20)      // pad_enable
            | (3u << 22)      // pad_interval: 16 DWORDs (one 32-elem row)
            | (3u << 25);     // pad_amount: 4 DWORDs (8 bf16) -> pitch 40
      g1[1] = ((unsigned)K & 0xFFFFu) << 16;        // tensor_dim0[15:0]
      g1[2] = ((unsigned)K >> 16) |                 // tensor_dim0[31:16]
              (((unsigned)M & 0xFFFFu) << 16);      // tensor_dim1[15:0]
      g1[3] = ((unsigned)M >> 16) | (32u << 16);    // tensor_dim1 hi | tile_dim0
      g1[4] = 128u;                                 // tile_dim1 (rows), tile_dim2=0
      g1[5] = (unsigned)K;                          // tensor_dim0_stride[31:0]
      g1[6] = 0u;                                   // stride hi | dim1_stride lo
      g1[7] = 0u;
      asm volatile("tensor_load_to_lds %0, %1" :: "s"(g0), "s"(g1) : "memory");
    }

    // prefetch next B tile chunk while TDM runs
    if (k0 + GBK < K) {
      int krow = tid >> 3;
      int nc   = (tid & 7) * 8;
      __builtin_prefetch(Bw + (size_t)(k0 + GBK + krow) * N + bn + nc, 0, 1);
    }

    // ---- stage B tile 32(K) x 64(N), transposed into Bs[n][k] ----
    {
      int krow = tid >> 3;          // 0..31
      int nc   = (tid & 7) * 8;
      v4u val = *(const v4u*)(Bw + (size_t)(k0 + krow) * N + bn + nc);
      unsigned short tmp[8];
      *(v4u*)tmp = val;
#pragma unroll
      for (int j = 0; j < 8; ++j) Bs[(nc + j) * APITCH + krow] = tmp[j];
    }
    if (wave == 0) __builtin_amdgcn_s_wait_tensorcnt(0);
    __syncthreads();

    // A fragments: lane=M row, elements K interleave {0-7,16-23}/{8-15,24-31}
    v16bf af[2], bfrag[2];
#pragma unroll
    for (int mi = 0; mi < 2; ++mi) {
      const unsigned short* p = &As[(wm * 32 + mi * 16 + l15) * APITCH + lh * 8];
      FragBF f; f.q[0] = *(const v4u*)p; f.q[1] = *(const v4u*)(p + 16);
      af[mi] = f.v;
    }
    // B fragments: lane=N col, elements K interleave {0-7,16-23}/{8-15,24-31}
#pragma unroll
    for (int ni = 0; ni < 2; ++ni) {
      const unsigned short* p = &Bs[(wn * 32 + ni * 16 + l15) * APITCH + lh * 8];
      FragBF f; f.q[0] = *(const v4u*)p; f.q[1] = *(const v4u*)(p + 16);
      bfrag[ni] = f.v;
    }
#pragma unroll
    for (int mi = 0; mi < 2; ++mi)
#pragma unroll
      for (int ni = 0; ni < 2; ++ni)
        acc[mi][ni] = __builtin_amdgcn_wmma_f32_16x16x32_bf16(
            false, af[mi], false, bfrag[ni], (short)0, acc[mi][ni], false, false);
    __syncthreads();
  }

  // epilogue: C layout lane=N, vgpr e -> M = e + 8*laneHalf
#pragma unroll
  for (int mi = 0; mi < 2; ++mi) {
#pragma unroll
    for (int ni = 0; ni < 2; ++ni) {
      int n = bn + wn * 32 + ni * 16 + l15;
      float bv = bias[n];
#pragma unroll
      for (int e = 0; e < 8; ++e) {
        int m = bm + wm * 32 + mi * 16 + e + lh * 8;
        float v = acc[mi][ni][e] + bv;
        if (GELU) v = 0.5f * v * (1.0f + erff(v * 0.70710678118f));
        size_t idx = (size_t)m * N + n;
        if (RESID) v += residual[idx];
        if (HASF) outF[idx] = v;
        if (HASB) outB[idx] = f2bf(v);
      }
    }
  }
}

// ---------------- flash attention (bf16 WMMA, online softmax) ----------------
#define VTPITCH 40

__global__ __launch_bounds__(128) void attn_kernel(
    const unsigned short* __restrict__ qkv,  // [M_ROWS, 3C] bf16 (q|k|v)
    const int* __restrict__ mask,            // [M_ROWS]
    unsigned short* __restrict__ yb) {       // [M_ROWS, C] bf16
  __shared__ unsigned short Vt[D_DIM * VTPITCH];   // V^T: [d][key] 64 x (32+pad)
  __shared__ unsigned short Pl[4 * 16 * VTPITCH];  // per-wave P: [m][key]

  const int tid  = threadIdx.x;
  const int lane = tid & 31;
  const int wave = tid >> 5;   // 4 waves, each owns 16 queries
  const int l15  = lane & 15;
  const int lh   = lane >> 4;
  const int qblk = blockIdx.x;
  const int h    = blockIdx.y;
  const int b    = blockIdx.z;
  const int q0   = qblk * 64 + wave * 16;

  // Q fragments (A-layout), held for the whole key loop
  v16bf qf[2];
#pragma unroll
  for (int d0 = 0; d0 < 2; ++d0) {
    const unsigned short* p = qkv + (size_t)(b * T_SEQ + q0 + l15) * (3 * C_DIM) +
                              h * D_DIM + d0 * 32 + lh * 8;
    FragBF f; f.q[0] = *(const v4u*)p; f.q[1] = *(const v4u*)(p + 16);
    qf[d0] = f.v;
  }

  int mq[8];
#pragma unroll
  for (int e = 0; e < 8; ++e) mq[e] = mask[b * T_SEQ + q0 + e + lh * 8];

  float mrow[8], lrow[8];
  v8f o[4];
#pragma unroll
  for (int e = 0; e < 8; ++e) { mrow[e] = -3.0e38f; lrow[e] = 0.0f; }
#pragma unroll
  for (int nt = 0; nt < 4; ++nt)
#pragma unroll
    for (int e = 0; e < 8; ++e) o[nt][e] = 0.0f;

  for (int kb = 0; kb < T_SEQ; kb += 32) {
    // stage V^T (32 keys x 64 d) cooperatively
#pragma unroll
    for (int it = 0; it < 2; ++it) {
      int c   = tid + it * 128;
      int key = c >> 3;
      int dc  = (c & 7) * 8;
      v4u val = *(const v4u*)(qkv + (size_t)(b * T_SEQ + kb + key) * (3 * C_DIM) +
                              2 * C_DIM + h * D_DIM + dc);
      unsigned short tmp[8];
      *(v4u*)tmp = val;
#pragma unroll
      for (int j = 0; j < 8; ++j) Vt[(dc + j) * VTPITCH + key] = tmp[j];
    }
    __syncthreads();

    // S = Q * K^T  (two 16-key tiles); K^T B-fragments straight from global
    v8f s[2];
#pragma unroll
    for (int nt = 0; nt < 2; ++nt)
#pragma unroll
      for (int e = 0; e < 8; ++e) s[nt][e] = 0.0f;

    int keyn[2], mk[2];
#pragma unroll
    for (int nt = 0; nt < 2; ++nt) {
      keyn[nt] = kb + nt * 16 + l15;
      mk[nt]   = mask[b * T_SEQ + keyn[nt]];
#pragma unroll
      for (int d0 = 0; d0 < 2; ++d0) {
        const unsigned short* p = qkv + (size_t)(b * T_SEQ + keyn[nt]) * (3 * C_DIM) +
                                  C_DIM + h * D_DIM + d0 * 32 + lh * 8;
        FragBF f; f.q[0] = *(const v4u*)p; f.q[1] = *(const v4u*)(p + 16);
        s[nt] = __builtin_amdgcn_wmma_f32_16x16x32_bf16(
            false, qf[d0], false, f.v, (short)0, s[nt], false, false);
      }
    }

    // online softmax: allowed = (mask[q] & mask[k]) | (k == 0)
#pragma unroll
    for (int e = 0; e < 8; ++e) {
      bool mqe = mq[e] != 0;
      bool a0 = (mqe && (mk[0] != 0)) || (keyn[0] == 0);
      bool a1 = (mqe && (mk[1] != 0));
      float s0 = s[0][e] * 0.125f;   // 1/sqrt(64)
      float s1 = s[1][e] * 0.125f;
      float mx = fmaxf(a0 ? s0 : -3.0e38f, a1 ? s1 : -3.0e38f);
#pragma unroll
      for (int off = 8; off >= 1; off >>= 1) mx = fmaxf(mx, __shfl_xor(mx, off, 16));
      float mnew  = fmaxf(mrow[e], mx);
      float alpha = __expf(mrow[e] - mnew);
      float p0 = a0 ? __expf(s0 - mnew) : 0.0f;
      float p1 = a1 ? __expf(s1 - mnew) : 0.0f;
      float rs = p0 + p1;
#pragma unroll
      for (int off = 8; off >= 1; off >>= 1) rs += __shfl_xor(rs, off, 16);
      lrow[e] = lrow[e] * alpha + rs;
      mrow[e] = mnew;
#pragma unroll
      for (int nt = 0; nt < 4; ++nt) o[nt][e] *= alpha;
      s[0][e] = p0; s[1][e] = p1;
    }

    // transpose P (C-layout -> A-layout) through LDS as bf16
    {
      unsigned short* pw = &Pl[wave * 16 * VTPITCH];
#pragma unroll
      for (int nt = 0; nt < 2; ++nt)
#pragma unroll
        for (int e = 0; e < 8; ++e)
          pw[(e + lh * 8) * VTPITCH + nt * 16 + l15] = f2bf(s[nt][e]);
    }
    __syncthreads();

    // O += P * V
    FragBF pf;
    {
      const unsigned short* pp = &Pl[wave * 16 * VTPITCH + l15 * VTPITCH + lh * 8];
      pf.q[0] = *(const v4u*)pp; pf.q[1] = *(const v4u*)(pp + 16);
    }
#pragma unroll
    for (int ntd = 0; ntd < 4; ++ntd) {
      const unsigned short* vp = &Vt[(ntd * 16 + l15) * VTPITCH + lh * 8];
      FragBF vf; vf.q[0] = *(const v4u*)vp; vf.q[1] = *(const v4u*)(vp + 16);
      o[ntd] = __builtin_amdgcn_wmma_f32_16x16x32_bf16(
          false, pf.v, false, vf.v, (short)0, o[ntd], false, false);
    }
    __syncthreads();
  }

  // normalize and scatter y (bf16)
#pragma unroll
  for (int e = 0; e < 8; ++e) {
    float inv = lrow[e] > 0.0f ? 1.0f / lrow[e] : 0.0f;
    int q = q0 + e + lh * 8;
#pragma unroll
    for (int ntd = 0; ntd < 4; ++ntd) {
      int d = ntd * 16 + l15;
      yb[(size_t)(b * T_SEQ + q) * C_DIM + h * D_DIM + d] = f2bf(o[ntd][e] * inv);
    }
  }
}

// ---------------- LayerNorm (one row per block) ----------------
template <int HASF, int HASB>
__global__ __launch_bounds__(256) void ln_kernel(
    const float* __restrict__ x, const float* __restrict__ w,
    const float* __restrict__ bparm, float* __restrict__ outF,
    unsigned short* __restrict__ outB) {
  __shared__ float red[8];
  const int row = blockIdx.x;
  const int tid = threadIdx.x;
  const float* xr = x + (size_t)row * C_DIM;

  float v[4];
  float sum = 0.0f;
#pragma unroll
  for (int i = 0; i < 4; ++i) { v[i] = xr[tid + i * 256]; sum += v[i]; }
#pragma unroll
  for (int off = 16; off >= 1; off >>= 1) sum += __shfl_xor(sum, off, 32);
  if ((tid & 31) == 0) red[tid >> 5] = sum;
  __syncthreads();
  float tot = 0.0f;
#pragma unroll
  for (int i = 0; i < 8; ++i) tot += red[i];
  float mean = tot * (1.0f / C_DIM);
  __syncthreads();

  float sq = 0.0f;
#pragma unroll
  for (int i = 0; i < 4; ++i) { float d = v[i] - mean; sq += d * d; }
#pragma unroll
  for (int off = 16; off >= 1; off >>= 1) sq += __shfl_xor(sq, off, 32);
  if ((tid & 31) == 0) red[tid >> 5] = sq;
  __syncthreads();
  float tot2 = 0.0f;
#pragma unroll
  for (int i = 0; i < 8; ++i) tot2 += red[i];
  float rstd = rsqrtf(tot2 * (1.0f / C_DIM) + LN_EPS);

#pragma unroll
  for (int i = 0; i < 4; ++i) {
    int c = tid + i * 256;
    float y = (v[i] - mean) * rstd * w[c] + bparm[c];
    size_t idx = (size_t)row * C_DIM + c;
    if (HASF) outF[idx] = y;
    if (HASB) outB[idx] = f2bf(y);
  }
}

// ---------------- launch ----------------
extern "C" void kernel_launch(void* const* d_in, const int* in_sizes, int n_in,
                              void* d_out, int out_size, void* d_ws, size_t ws_size,
                              hipStream_t stream) {
  (void)in_sizes; (void)n_in; (void)out_size; (void)ws_size;
  const float* x      = (const float*)d_in[0];
  const int*   mask   = (const int*)  d_in[1];
  const float* attn_w = (const float*)d_in[2];
  const float* attn_b = (const float*)d_in[3];
  const float* proj_w = (const float*)d_in[4];
  const float* proj_b = (const float*)d_in[5];
  const float* ln1_w  = (const float*)d_in[6];
  const float* ln1_b  = (const float*)d_in[7];
  const float* lin1_w = (const float*)d_in[8];
  const float* lin1_b = (const float*)d_in[9];
  const float* lin2_w = (const float*)d_in[10];
  const float* lin2_b = (const float*)d_in[11];
  const float* ln2_w  = (const float*)d_in[12];
  const float* ln2_b  = (const float*)d_in[13];

  char* ws = (char*)d_ws;
  size_t off = 0;
  auto alloc = [&](size_t bytes) {
    size_t o = off; off += (bytes + 255) & ~(size_t)255; return o;
  };

  unsigned short* xb     = (unsigned short*)(ws + alloc((size_t)M_ROWS * C_DIM * 2));
  unsigned short* wqkvb  = (unsigned short*)(ws + alloc((size_t)C_DIM * 3 * C_DIM * 2));
  unsigned short* wprojb = (unsigned short*)(ws + alloc((size_t)C_DIM * C_DIM * 2));
  unsigned short* w1b    = (unsigned short*)(ws + alloc((size_t)C_DIM * HID_DIM * 2));
  unsigned short* w2b    = (unsigned short*)(ws + alloc((size_t)HID_DIM * C_DIM * 2));
  // big region: qkv (24MB) + y (8MB); reused afterwards as ff1 hidden (32MB)
  size_t bigOff = alloc((size_t)M_ROWS * HID_DIM * 2);
  unsigned short* qkvb = (unsigned short*)(ws + bigOff);
  unsigned short* yb   = (unsigned short*)(ws + bigOff + (size_t)M_ROWS * 3 * C_DIM * 2);
  unsigned short* ff1b = (unsigned short*)(ws + bigOff);
  float* h1f = (float*)(ws + alloc((size_t)M_ROWS * C_DIM * 4));  // reused as h2f
  float* hf  = (float*)(ws + alloc((size_t)M_ROWS * C_DIM * 4));
  unsigned short* hb = (unsigned short*)(ws + alloc((size_t)M_ROWS * C_DIM * 2));

  // convert activations + weights to bf16
  cvt_kernel<<<dim3(2048), dim3(256), 0, stream>>>(x, xb, M_ROWS * C_DIM);
  cvt_kernel<<<dim3(2048), dim3(256), 0, stream>>>(attn_w, wqkvb, C_DIM * 3 * C_DIM);
  cvt_kernel<<<dim3(1024), dim3(256), 0, stream>>>(proj_w, wprojb, C_DIM * C_DIM);
  cvt_kernel<<<dim3(2048), dim3(256), 0, stream>>>(lin1_w, w1b, C_DIM * HID_DIM);
  cvt_kernel<<<dim3(2048), dim3(256), 0, stream>>>(lin2_w, w2b, HID_DIM * C_DIM);

  // fused QKV projection (bf16 out)
  gemm_bf16_kernel<0, 0, 1, 0>
      <<<dim3(3 * C_DIM / GBN, M_ROWS / GBM), dim3(256), 0, stream>>>(
          xb, wqkvb, M_ROWS, 3 * C_DIM, C_DIM, attn_b, nullptr, nullptr, qkvb);

  // flash attention
  attn_kernel<<<dim3(T_SEQ / 64, H_NUM, B_DIM), dim3(128), 0, stream>>>(qkvb, mask, yb);

  // output projection + residual x (fp32 out)
  gemm_bf16_kernel<1, 1, 0, 0>
      <<<dim3(C_DIM / GBN, M_ROWS / GBM), dim3(256), 0, stream>>>(
          yb, wprojb, M_ROWS, C_DIM, C_DIM, proj_b, x, h1f, nullptr);

  // post-LN 1 -> h (fp32 for residual, bf16 for GEMM A)
  ln_kernel<1, 1><<<dim3(M_ROWS), dim3(256), 0, stream>>>(h1f, ln1_w, ln1_b, hf, hb);

  // MLP up + exact GELU (bf16 out)
  gemm_bf16_kernel<0, 0, 1, 1>
      <<<dim3(HID_DIM / GBN, M_ROWS / GBM), dim3(256), 0, stream>>>(
          hb, w1b, M_ROWS, HID_DIM, C_DIM, lin1_b, nullptr, nullptr, ff1b);

  // MLP down + residual h (fp32 out, reuses h1f)
  gemm_bf16_kernel<1, 1, 0, 0>
      <<<dim3(C_DIM / GBN, M_ROWS / GBM), dim3(256), 0, stream>>>(
          ff1b, w2b, M_ROWS, C_DIM, HID_DIM, lin2_b, hf, h1f, nullptr);

  // post-LN 2 -> final output
  ln_kernel<1, 0><<<dim3(M_ROWS), dim3(256), 0, stream>>>(
      h1f, ln2_w, ln2_b, (float*)d_out, nullptr);
}
